// LongformerSelfAttention_43894565765633
// MI455X (gfx1250) — matrix operand order repaired and checked by reading, not compiled, not run on hardware
//
#include <hip/hip_runtime.h>

// ---------------- types & helpers ----------------
typedef __attribute__((ext_vector_type(16))) __bf16 v16bf;
typedef __attribute__((ext_vector_type(8)))  float  v8f;
typedef __attribute__((ext_vector_type(8)))  __bf16 bf16x8;

union V16U { v16bf v; bf16x8 h[2]; };
union U4BF { uint4 u; __bf16 b[8]; };

#define BB   2
#define SS   4096
#define DD   768
#define HH   12
#define HD   64
#define WW   256
#define GG   32
#define SCALE 0.125f
#define NEGV (-1.0e9f)

#if __has_builtin(__builtin_amdgcn_global_load_async_to_lds_b128) && \
    __has_builtin(__builtin_amdgcn_s_wait_asynccnt)
#define HAVE_ASYNC_LDS 1
#else
#define HAVE_ASYNC_LDS 0
#endif

#if HAVE_ASYNC_LDS
typedef int v4i __attribute__((vector_size(16)));
__device__ __forceinline__ void async_cp_row128(const void* gsrc, void* lds) {
    // copy 128 bytes: 8 x b128 async global->LDS (per-lane addresses)
#pragma unroll
    for (int j = 0; j < 8; ++j) {
        __builtin_amdgcn_global_load_async_to_lds_b128(
            (v4i*)((const char*)gsrc + j * 16),
            (v4i*)((char*)lds + j * 16), 0, 0);
    }
}
#endif

__device__ __forceinline__ __bf16 f2bf(float f) { return (__bf16)f; } // hw cvt

// A-matrix fragment (16x32 bf16), row fixed per lane, from row-major LDS.
// lane l: row = l%16 ; element j -> K = (j%8) + 8*(2*(j/8) + l/16)
__device__ __forceinline__ v16bf frag_a(const __bf16* rowbase, int k0, int half) {
    V16U r;
    r.h[0] = *(const bf16x8*)(rowbase + k0 + 8  * half);
    r.h[1] = *(const bf16x8*)(rowbase + k0 + 16 + 8 * half);
    return r.v;
}
// B-matrix fragment (32x16 bf16) from LDS stored as Bt[N][K]:
// lane l: col = l%16 ; element j -> K = (j%8) + 8*(2*(j/8) + l/16)
__device__ __forceinline__ v16bf frag_b(const __bf16* colbase, int k0, int half) {
    V16U r;
    r.h[0] = *(const bf16x8*)(colbase + k0 + 8  * half);
    r.h[1] = *(const bf16x8*)(colbase + k0 + 16 + 8 * half);
    return r.v;
}
__device__ __forceinline__ v8f v8zero() {
    v8f z;
#pragma unroll
    for (int e = 0; e < 8; ++e) z[e] = 0.0f;
    return z;
}

// ---------------- projection GEMM: out = (X @ W + b) * scale ----------------
// X: (B*S=8192, 768) f32 ; W: (768,768) f32 ; out: bf16 in (B,H,S,HD) layout
__global__ __launch_bounds__(256) void gemm_proj(
    const float* __restrict__ X, const float* __restrict__ Wm,
    const float* __restrict__ bias, __bf16* __restrict__ out, float scale)
{
    __shared__ __bf16 As[128 * 32];     // A tile, row-major [row][k]
    __shared__ __bf16 Bts[128 * 32];    // B tile, transposed [n][k]

    const int t = threadIdx.x;
    const int lane = t & 31, wv = t >> 5;
    const int l15 = lane & 15, half = lane >> 4;
    const int bm = blockIdx.x * 128, bn = blockIdx.y * 128;
    const int wm = (wv >> 1) * 32, wn = (wv & 1) * 64;

    v8f acc[2][4];
#pragma unroll
    for (int mt = 0; mt < 2; ++mt)
#pragma unroll
        for (int nt = 0; nt < 4; ++nt) acc[mt][nt] = v8zero();

    const int arow = t >> 1;            // 0..127
    const int ac0  = (t & 1) * 16;
    const int bk   = t & 31;            // 0..31
    const int bn0  = (t >> 5) * 16;     // 0..112

    for (int kb = 0; kb < 24; ++kb) {
        const int k0 = kb * 32;
        const float* xs = X + (size_t)(bm + arow) * DD + k0 + ac0;
        const float* wsrc = Wm + (size_t)(k0 + bk) * DD + bn + bn0;
        if (kb < 23) {  // prefetch next K-step tiles (global_prefetch_b8)
            __builtin_prefetch(xs + 32, 0, 1);
            __builtin_prefetch(wsrc + 32 * DD, 0, 1);
        }
#pragma unroll
        for (int j = 0; j < 16; ++j) As[arow * 32 + ac0 + j] = f2bf(xs[j]);
#pragma unroll
        for (int j = 0; j < 16; ++j) Bts[(bn0 + j) * 32 + bk] = f2bf(wsrc[j]);
        __syncthreads();

#pragma unroll
        for (int mt = 0; mt < 2; ++mt) {
            v16bf a = frag_a(&As[(wm + mt * 16 + l15) * 32], 0, half);
#pragma unroll
            for (int nt = 0; nt < 4; ++nt) {
                v16bf b = frag_b(&Bts[(wn + nt * 16 + l15) * 32], 0, half);
                acc[mt][nt] = __builtin_amdgcn_wmma_f32_16x16x32_bf16(
                    false, a, false, b, (short)0, acc[mt][nt], false, false);
            }
        }
        __syncthreads();
    }

#pragma unroll
    for (int mt = 0; mt < 2; ++mt)
#pragma unroll
        for (int nt = 0; nt < 4; ++nt)
#pragma unroll
            for (int i = 0; i < 8; ++i) {
                int m = bm + wm + mt * 16 + i + 8 * half;     // flat B*S row
                int n = bn + wn + nt * 16 + l15;              // 0..767
                float v = (acc[mt][nt][i] + bias[n]) * scale;
                int b = m >> 12, s = m & (SS - 1);
                int h = n >> 6,  d = n & (HD - 1);
                out[((((size_t)b * HH + h) * SS) + s) * HD + d] = f2bf(v);
            }
}

// ---------------- flash attention (band + global-token variants) ----------------
template<bool BAND>
__global__ __launch_bounds__(32) void flash_attn(
    const __bf16* __restrict__ Q, const __bf16* __restrict__ K,
    const __bf16* __restrict__ V, float* __restrict__ O)
{
    __shared__ __bf16 Qs[32 * 64];
    __shared__ __bf16 Ks[32 * 64];     // [key][d]
    __shared__ __bf16 Vts[64 * 32];    // [d][key]
    __shared__ __bf16 Ps[32 * 32];     // probs bf16 [q][key]
    __shared__ float mrow[32], lrow[32], frow[32], cmax[32], psum[32], linv[32];

    const int lane = threadIdx.x & 31;
    const int l15 = lane & 15, half = lane >> 4;

    int bh, qbase, c = 0;
    if (BAND) {
        int id = blockIdx.x;
        int sub = id & 7; c = (id >> 3) & 15; bh = id >> 7;
        qbase = c * WW + sub * 32;
    } else { bh = blockIdx.x; qbase = 0; }

    const __bf16* Qh = Q + (size_t)bh * SS * HD;
    const __bf16* Kh = K + (size_t)bh * SS * HD;
    const __bf16* Vh = V + (size_t)bh * SS * HD;

    // load 32 query rows (128B each): async DMA to LDS when available
#if HAVE_ASYNC_LDS
    async_cp_row128(Qh + (size_t)(qbase + lane) * HD, &Qs[lane * 64]);
    __builtin_amdgcn_s_wait_asynccnt(0);
#else
    {
        const uint4* src = (const uint4*)(Qh + (size_t)(qbase + lane) * HD);
        uint4* dst = (uint4*)(&Qs[lane * 64]);
#pragma unroll
        for (int j = 0; j < 8; ++j) dst[j] = src[j];
    }
#endif
    mrow[lane] = -1e30f; lrow[lane] = 0.0f;

    v8f oacc[2][4];
#pragma unroll
    for (int mt = 0; mt < 2; ++mt)
#pragma unroll
        for (int nt = 0; nt < 4; ++nt) oacc[mt][nt] = v8zero();
    __syncthreads();

    const int nch = BAND ? 25 : (SS / 32);
    for (int ci = 0; ci < nch; ++ci) {
        int kstart; bool masked;
        if (BAND) {
            if (ci == 0) { kstart = 0; masked = false; }   // global-key chunk
            else {
                kstart = c * WW - WW + (ci - 1) * 32; masked = true;
                if (kstart + 31 < qbase - WW || kstart > qbase + 31 + WW ||
                    kstart + 31 < 0 || kstart >= SS) continue;
            }
        } else { kstart = ci * 32; masked = false; }

        // stage K (row-major, async DMA) and V (register transpose).
        // OOB rows: clamp address; mask forces those scores to NEG -> P == 0,
        // so clamped (finite) K/V data contributes nothing.
        {
            int kabs = kstart + lane;
            int kcl = kabs < 0 ? 0 : (kabs >= SS ? SS - 1 : kabs);
#if HAVE_ASYNC_LDS
            async_cp_row128(Kh + (size_t)kcl * HD, &Ks[lane * 64]);
#else
            const uint4* ksrc = (const uint4*)(Kh + (size_t)kcl * HD);
            uint4* kdst = (uint4*)(&Ks[lane * 64]);
#pragma unroll
            for (int j = 0; j < 8; ++j) kdst[j] = ksrc[j];
#endif
            const uint4* vsrc = (const uint4*)(Vh + (size_t)kcl * HD);
#pragma unroll
            for (int j = 0; j < 8; ++j) {
                U4BF u; u.u = vsrc[j];
#pragma unroll
                for (int e = 0; e < 8; ++e) Vts[(j * 8 + e) * 32 + lane] = u.b[e];
            }
#if HAVE_ASYNC_LDS
            __builtin_amdgcn_s_wait_asynccnt(0);
#endif
        }
        __syncthreads();

        // scores: S = Q(32x64) @ K^T  -> 2x2 tiles
        v8f sacc[2][2];
#pragma unroll
        for (int mt = 0; mt < 2; ++mt) {
#pragma unroll
            for (int nk = 0; nk < 2; ++nk) sacc[mt][nk] = v8zero();
#pragma unroll
            for (int kd = 0; kd < 64; kd += 32) {
                v16bf a = frag_a(&Qs[(mt * 16 + l15) * 64], kd, half);
#pragma unroll
                for (int nk = 0; nk < 2; ++nk) {
                    v16bf b = frag_b(&Ks[(nk * 16 + l15) * 64], kd, half);
                    sacc[mt][nk] = __builtin_amdgcn_wmma_f32_16x16x32_bf16(
                        false, a, false, b, (short)0, sacc[mt][nk], false, false);
                }
            }
        }

        // mask + per-row chunk max (16-lane butterfly)
#pragma unroll
        for (int mt = 0; mt < 2; ++mt)
#pragma unroll
            for (int i = 0; i < 8; ++i) {
                int rl = mt * 16 + i + 8 * half;
                int qabs = qbase + rl;
                float mx = -1e30f;
#pragma unroll
                for (int nk = 0; nk < 2; ++nk) {
                    int kabs = kstart + nk * 16 + l15;
                    float s = sacc[mt][nk][i];
                    bool valid = !masked ||
                        (kabs >= 0 && kabs < SS && kabs >= qabs - WW && kabs <= qabs + WW);
                    s = valid ? s : NEGV;
                    sacc[mt][nk][i] = s;
                    mx = fmaxf(mx, s);
                }
#pragma unroll
                for (int sh = 1; sh < 16; sh <<= 1)
                    mx = fmaxf(mx, __shfl_xor(mx, sh, 16));
                if (l15 == 0) cmax[rl] = mx;
            }
        __syncthreads();
        { // online-softmax row-state update (thread t owns row t)
            float mo = mrow[lane];
            float mn = fmaxf(mo, cmax[lane]);
            frow[lane] = __expf(mo - mn);
            mrow[lane] = mn;
        }
        __syncthreads();

        // P = exp(S - m), store bf16, per-row partial sum
#pragma unroll
        for (int mt = 0; mt < 2; ++mt)
#pragma unroll
            for (int i = 0; i < 8; ++i) {
                int rl = mt * 16 + i + 8 * half;
                float mn = mrow[rl];
                float ps = 0.0f;
#pragma unroll
                for (int nk = 0; nk < 2; ++nk) {
                    float p = __expf(sacc[mt][nk][i] - mn);
                    ps += p;
                    Ps[rl * 32 + nk * 16 + l15] = f2bf(p);
                }
#pragma unroll
                for (int sh = 1; sh < 16; sh <<= 1)
                    ps += __shfl_xor(ps, sh, 16);
                if (l15 == 0) psum[rl] = ps;
            }
        __syncthreads();
        lrow[lane] = lrow[lane] * frow[lane] + psum[lane];

        // rescale running output, then O += P @ Vchunk
#pragma unroll
        for (int mt = 0; mt < 2; ++mt)
#pragma unroll
            for (int nt = 0; nt < 4; ++nt)
#pragma unroll
                for (int i = 0; i < 8; ++i) {
                    int rl = mt * 16 + i + 8 * half;
                    oacc[mt][nt][i] *= frow[rl];
                }
#pragma unroll
        for (int mt = 0; mt < 2; ++mt) {
            v16bf a = frag_a(&Ps[(mt * 16 + l15) * 32], 0, half);
#pragma unroll
            for (int nt = 0; nt < 4; ++nt) {
                v16bf b = frag_b(&Vts[(nt * 16 + l15) * 32], 0, half);
                oacc[mt][nt] = __builtin_amdgcn_wmma_f32_16x16x32_bf16(
                    false, a, false, b, (short)0, oacc[mt][nt], false, false);
            }
        }
        __syncthreads();
    }

    linv[lane] = 1.0f / lrow[lane];
    __syncthreads();
    float* Oh = BAND ? (O + ((size_t)bh * SS + qbase) * HD)
                     : (O + (size_t)bh * GG * HD);
#pragma unroll
    for (int mt = 0; mt < 2; ++mt)
#pragma unroll
        for (int nt = 0; nt < 4; ++nt)
#pragma unroll
            for (int i = 0; i < 8; ++i) {
                int rl = mt * 16 + i + 8 * half;
                int n  = nt * 16 + l15;
                Oh[(size_t)rl * HD + n] = oacc[mt][nt][i] * linv[rl];
            }
}

// ---------------- final assembly: (B,H,S,HD) -> (B,S,D), rows<G from gout ----------------
__global__ __launch_bounds__(256) void assemble(
    const float* __restrict__ bandO, const float* __restrict__ gO,
    float* __restrict__ out)
{
    size_t idx = (size_t)blockIdx.x * 256 + threadIdx.x;
    int n = (int)(idx % DD);
    size_t bs = idx / DD;
    int s = (int)(bs % SS);
    int b = (int)(bs / SS);
    int h = n >> 6, d = n & (HD - 1);
    float v;
    if (s < GG) v = gO[(((size_t)b * HH + h) * GG + s) * HD + d];
    else        v = bandO[(((size_t)b * HH + h) * SS + s) * HD + d];
    out[idx] = v;
}

// ---------------- launcher ----------------
extern "C" void kernel_launch(void* const* d_in, const int* in_sizes, int n_in,
                              void* d_out, int out_size, void* d_ws, size_t ws_size,
                              hipStream_t stream) {
    const float* hs  = (const float*)d_in[0];
    const float* Wq  = (const float*)d_in[1];  const float* bq  = (const float*)d_in[2];
    const float* Wk  = (const float*)d_in[3];  const float* bk  = (const float*)d_in[4];
    const float* Wv  = (const float*)d_in[5];  const float* bv  = (const float*)d_in[6];
    const float* Wqg = (const float*)d_in[7];  const float* bqg = (const float*)d_in[8];
    const float* Wkg = (const float*)d_in[9];  const float* bkg = (const float*)d_in[10];
    const float* Wvg = (const float*)d_in[11]; const float* bvg = (const float*)d_in[12];

    char* ws = (char*)d_ws;
    const size_t bfsz = (size_t)BB * HH * SS * HD * sizeof(unsigned short); // 12.58 MB
    __bf16* q  = (__bf16*)(ws + 0 * bfsz);
    __bf16* k  = (__bf16*)(ws + 1 * bfsz);
    __bf16* v  = (__bf16*)(ws + 2 * bfsz);
    __bf16* qg = (__bf16*)(ws + 3 * bfsz);
    __bf16* kg = (__bf16*)(ws + 4 * bfsz);
    __bf16* vg = (__bf16*)(ws + 5 * bfsz);
    float* bandO = (float*)(ws + 6 * bfsz);
    float* gO    = (float*)(ws + 6 * bfsz + (size_t)BB * HH * SS * HD * sizeof(float));

    dim3 gg(64, 6, 1);
    gemm_proj<<<gg, 256, 0, stream>>>(hs, Wq,  bq,  q,  SCALE);
    gemm_proj<<<gg, 256, 0, stream>>>(hs, Wk,  bk,  k,  1.0f);
    gemm_proj<<<gg, 256, 0, stream>>>(hs, Wv,  bv,  v,  1.0f);
    gemm_proj<<<gg, 256, 0, stream>>>(hs, Wqg, bqg, qg, SCALE);
    gemm_proj<<<gg, 256, 0, stream>>>(hs, Wkg, bkg, kg, 1.0f);
    gemm_proj<<<gg, 256, 0, stream>>>(hs, Wvg, bvg, vg, 1.0f);

    flash_attn<true><<<BB * HH * (SS / WW) * (WW / 32), 32, 0, stream>>>(q, k, v, bandO);
    flash_attn<false><<<BB * HH, 32, 0, stream>>>(qg, kg, vg, gO);

    assemble<<<(BB * SS * DD) / 256, 256, 0, stream>>>(bandO, gO, (float*)d_out);
}
